// AtomAttentionEncoder_317827580849
// MI455X (gfx1250) — compile-verified, matrix operand order checked by reading, not run on hardware
//
#include <hip/hip_runtime.h>

// ---------------- problem constants ----------------
#define NATOM  8192
#define DMODEL 128
#define WWIN   32
#define HKEY   128
#define NHEAD  4
#define DHEAD  32
#define NWIN   256
#define NLAYER 3
#define NTOK   2048
#define FTOK   768

// ---------------- WMMA tiling ----------------
#define BM 128
#define BN 64
#define KC 32
#define PITCH 40   // bf16 elements per LDS row: 80B, 16B-aligned, conflict-free

typedef __attribute__((ext_vector_type(16))) __bf16 v16bf;
typedef __attribute__((ext_vector_type(8)))  float  v8f;

enum { ACT_NONE = 0, ACT_SIGMOID = 1, ACT_RELU = 2, ACT_GATED = 3 };
enum { DUAL_ADALN = 0, DUAL_SWIGLU = 1 };

#if defined(__HIP_DEVICE_COMPILE__) && __has_builtin(__builtin_amdgcn_global_load_async_to_lds_b32)
#define HAVE_ASYNC_LDS 1
typedef __attribute__((address_space(1))) int gint_t;   // global (printed '__device__' by clang)
typedef __attribute__((address_space(3))) int lint_t;   // LDS
#endif

union FragU { v16bf v; uint4 u[2]; };

__device__ __forceinline__ float sigmoidf_dev(float x) { return 1.0f / (1.0f + __expf(-x)); }

// Load a 16x32 bf16 A/B fragment from an LDS tile stored [row][PITCH].
// lane L (0..15): row = row0 + L, K 0..7 and 16..23 ; lane 16+L: K 8..15 and 24..31.
__device__ __forceinline__ v16bf load_frag(const __bf16* base, int row0, int pitch) {
  const int lane = threadIdx.x & 31;
  const __bf16* p = base + (row0 + (lane & 15)) * pitch + ((lane >> 4) << 3);
  FragU f;
  f.u[0] = *(const uint4*)(p);
  f.u[1] = *(const uint4*)(p + 16);
  return f.v;
}

// ---------------- a = q + r @ r_to_q_w ----------------
__global__ __launch_bounds__(256) void k_init_a(const float* __restrict__ q,
                                                const float* __restrict__ r,
                                                const float* __restrict__ w,
                                                float* __restrict__ a) {
  const int idx = blockIdx.x * 256 + threadIdx.x;
  const int n = idx >> 7, d = idx & 127;
  a[idx] = q[idx] + r[n * 3 + 0] * w[d] + r[n * 3 + 1] * w[DMODEL + d] +
           r[n * 3 + 2] * w[2 * DMODEL + d];
}

// ---------------- row LayerNorm (width 128), one wave per row ----------------
__global__ __launch_bounds__(128) void k_ln(const float* __restrict__ x,
                                            float* __restrict__ y) {
  const int row  = blockIdx.x * 4 + (threadIdx.x >> 5);
  const int lane = threadIdx.x & 31;
  const float4 v = *(const float4*)(x + (size_t)row * DMODEL + lane * 4);
  float s = v.x + v.y + v.z + v.w;
  float q = v.x * v.x + v.y * v.y + v.z * v.z + v.w * v.w;
  for (int m = 16; m; m >>= 1) { s += __shfl_xor(s, m); q += __shfl_xor(q, m); }
  const float mean = s * (1.0f / 128.0f);
  const float var  = q * (1.0f / 128.0f) - mean * mean;
  const float rs   = rsqrtf(var + 1e-5f);
  float4 o;
  o.x = (v.x - mean) * rs; o.y = (v.y - mean) * rs;
  o.z = (v.z - mean) * rs; o.w = (v.w - mean) * rs;
  *(float4*)(y + (size_t)row * DMODEL + lane * 4) = o;
}

// ---------------- f32 -> bf16 weight conversion ----------------
__global__ __launch_bounds__(256) void k_cvt(const float* __restrict__ s,
                                             __bf16* __restrict__ d, int n) {
  const int i = blockIdx.x * 256 + threadIdx.x;
  if (i < n) d[i] = (__bf16)s[i];
}

// ---------------- generic bf16 WMMA GEMM with fused epilogues ----------------
// C[M,Nn] = act(A[M,K] @ Wt[K,Nn] + bias) ; optional row gather on A.
// K templated (128/256): k-loop fully unrolls, addressing folds to constants.
template <int K, bool GATHER>
__global__ __launch_bounds__(256) void k_gemm(
    const float* __restrict__ A, const int* __restrict__ rowidx,
    const __bf16* __restrict__ Wt, const float* __restrict__ bias,
    float* __restrict__ Cout, const float* __restrict__ gate,
    float* __restrict__ accdst, int Nn, int mode) {
  __shared__ __bf16 sA[BM * PITCH];
  __shared__ __bf16 sW[BN * PITCH];
  const int tid  = threadIdx.x;
  const int bm   = blockIdx.x * BM;
  const int bn   = blockIdx.y * BN;
  const int wave = tid >> 5;
  const int wm   = (wave >> 1) * 32;
  const int wn   = (wave & 1) * 32;
  // hoisted per-slot A addressing (rows independent of k0)
  const float* aptr[4];
  __bf16*      sptr[4];
#pragma unroll
  for (int it = 0; it < 4; ++it) {
    const int slot = it * 256 + tid;
    const int rrow = slot >> 3;
    const int c4   = (slot & 7) << 2;
    const int grow = bm + rrow;
    const int arow = GATHER ? rowidx[grow] : grow;
    aptr[it] = A + (size_t)arow * K + c4;
    sptr[it] = sA + rrow * PITCH + c4;
  }
  const __bf16* wptr = Wt + (size_t)(tid >> 3) * Nn + bn + ((tid & 7) << 3);
  const int kkofs = tid >> 3;           // k row this thread writes for W tile
  const int ngofs = (tid & 7) << 3;
  v8f acc[2][2] = {};
#pragma unroll
  for (int k0 = 0; k0 < K; k0 += KC) {
#pragma unroll
    for (int it = 0; it < 4; ++it) {    // A tile: 128 rows x 32 cols f32 -> bf16
      if (k0 + KC < K) __builtin_prefetch(aptr[it] + k0 + KC, 0, 3);  // folds at compile time
      const float4 f = *(const float4*)(aptr[it] + k0);
      __bf16* d = sptr[it];
      d[0] = (__bf16)f.x; d[1] = (__bf16)f.y; d[2] = (__bf16)f.z; d[3] = (__bf16)f.w;
    }
    {                                   // W tile: 32(k) x 64(n), store transposed [n][k]
      union { uint4 u; __bf16 h[8]; } wrow;
      wrow.u = *(const uint4*)(wptr + (size_t)k0 * Nn);
#pragma unroll
      for (int j = 0; j < 8; ++j) sW[(ngofs + j) * PITCH + kkofs] = wrow.h[j];
    }
    __syncthreads();
    v16bf bfr0 = load_frag(sW, wn, PITCH);
    v16bf bfr1 = load_frag(sW, wn + 16, PITCH);
#pragma unroll
    for (int mi = 0; mi < 2; ++mi) {
      v16bf af = load_frag(sA, wm + mi * 16, PITCH);
      acc[mi][0] = __builtin_amdgcn_wmma_f32_16x16x32_bf16(false, af, false, bfr0,
                                                           (short)0, acc[mi][0], false, false);
      acc[mi][1] = __builtin_amdgcn_wmma_f32_16x16x32_bf16(false, af, false, bfr1,
                                                           (short)0, acc[mi][1], false, false);
    }
    __syncthreads();
  }
  const int lane = tid & 31;
  const int cn   = lane & 15;
  const int mhi  = (lane >> 4) << 3;
#pragma unroll
  for (int mi = 0; mi < 2; ++mi)
#pragma unroll
    for (int ni = 0; ni < 2; ++ni) {
      const int col = bn + wn + ni * 16 + cn;
      const float bv = bias ? bias[col] : 0.0f;
#pragma unroll
      for (int v = 0; v < 8; ++v) {
        const int row = bm + wm + mi * 16 + mhi + v;
        const size_t off = (size_t)row * Nn + col;
        const float x = acc[mi][ni][v] + bv;
        if (mode == ACT_NONE)          Cout[off] = x;
        else if (mode == ACT_SIGMOID)  Cout[off] = sigmoidf_dev(x);
        else if (mode == ACT_RELU)     Cout[off] = fmaxf(x, 0.0f);
        else                           accdst[off] += gate[off] * x;  // gated residual add
      }
    }
}

// ---------------- dual-weight GEMM: AdaLN combine / SwiGLU combine ----------------
template <int K>
__global__ __launch_bounds__(256) void k_gemm_dual(
    const float* __restrict__ A, const __bf16* __restrict__ W1t,
    const __bf16* __restrict__ W2t, const float* __restrict__ ln,
    float* __restrict__ Out, int Nn, int mode) {
  __shared__ __bf16 sA[BM * PITCH];
  __shared__ __bf16 sW1[BN * PITCH];
  __shared__ __bf16 sW2[BN * PITCH];
  const int tid  = threadIdx.x;
  const int bm   = blockIdx.x * BM;
  const int bn   = blockIdx.y * BN;
  const int wave = tid >> 5;
  const int wm   = (wave >> 1) * 32;
  const int wn   = (wave & 1) * 32;
  const float* aptr[4];
  __bf16*      sptr[4];
#pragma unroll
  for (int it = 0; it < 4; ++it) {
    const int slot = it * 256 + tid;
    const int rrow = slot >> 3;
    const int c4   = (slot & 7) << 2;
    aptr[it] = A + (size_t)(bm + rrow) * K + c4;
    sptr[it] = sA + rrow * PITCH + c4;
  }
  const size_t wofs = (size_t)(tid >> 3) * Nn + bn + ((tid & 7) << 3);
  const int kkofs = tid >> 3;
  const int ngofs = (tid & 7) << 3;
  v8f acc1[2][2] = {}, acc2[2][2] = {};
#pragma unroll
  for (int k0 = 0; k0 < K; k0 += KC) {
#pragma unroll
    for (int it = 0; it < 4; ++it) {
      if (k0 + KC < K) __builtin_prefetch(aptr[it] + k0 + KC, 0, 3);
      const float4 f = *(const float4*)(aptr[it] + k0);
      __bf16* d = sptr[it];
      d[0] = (__bf16)f.x; d[1] = (__bf16)f.y; d[2] = (__bf16)f.z; d[3] = (__bf16)f.w;
    }
    {
      union { uint4 u; __bf16 h[8]; } w1, w2;
      w1.u = *(const uint4*)(W1t + wofs + (size_t)k0 * Nn);
      w2.u = *(const uint4*)(W2t + wofs + (size_t)k0 * Nn);
#pragma unroll
      for (int j = 0; j < 8; ++j) {
        sW1[(ngofs + j) * PITCH + kkofs] = w1.h[j];
        sW2[(ngofs + j) * PITCH + kkofs] = w2.h[j];
      }
    }
    __syncthreads();
    v16bf b10 = load_frag(sW1, wn, PITCH);
    v16bf b11 = load_frag(sW1, wn + 16, PITCH);
    v16bf b20 = load_frag(sW2, wn, PITCH);
    v16bf b21 = load_frag(sW2, wn + 16, PITCH);
#pragma unroll
    for (int mi = 0; mi < 2; ++mi) {
      v16bf af = load_frag(sA, wm + mi * 16, PITCH);
      acc1[mi][0] = __builtin_amdgcn_wmma_f32_16x16x32_bf16(false, af, false, b10,
                                                            (short)0, acc1[mi][0], false, false);
      acc1[mi][1] = __builtin_amdgcn_wmma_f32_16x16x32_bf16(false, af, false, b11,
                                                            (short)0, acc1[mi][1], false, false);
      acc2[mi][0] = __builtin_amdgcn_wmma_f32_16x16x32_bf16(false, af, false, b20,
                                                            (short)0, acc2[mi][0], false, false);
      acc2[mi][1] = __builtin_amdgcn_wmma_f32_16x16x32_bf16(false, af, false, b21,
                                                            (short)0, acc2[mi][1], false, false);
    }
    __syncthreads();
  }
  const int lane = tid & 31;
  const int cn   = lane & 15;
  const int mhi  = (lane >> 4) << 3;
#pragma unroll
  for (int mi = 0; mi < 2; ++mi)
#pragma unroll
    for (int ni = 0; ni < 2; ++ni) {
      const int col = bn + wn + ni * 16 + cn;
#pragma unroll
      for (int v = 0; v < 8; ++v) {
        const int row = bm + wm + mi * 16 + mhi + v;
        const size_t off = (size_t)row * Nn + col;
        const float x1 = acc1[mi][ni][v];
        const float x2 = acc2[mi][ni][v];
        if (mode == DUAL_ADALN)
          Out[off] = sigmoidf_dev(x1) * ln[off] + x2;        // b = sig(cn@S)*ln(a)+cn@Sh
        else
          Out[off] = x1 * sigmoidf_dev(x1) * x2;             // h = silu(g)*lin
      }
    }
}

// ---------------- fused windowed attention: one block per (window, head) ----------------
__global__ __launch_bounds__(64) void k_attn(
    const float* __restrict__ qh, const float* __restrict__ kh,
    const float* __restrict__ vh, const float* __restrict__ g,
    const float* __restrict__ bias, const int* __restrict__ kidx,
    const unsigned char* __restrict__ pad, float* __restrict__ o) {
  __shared__ __bf16 sQ[WWIN * PITCH];
  __shared__ __bf16 sK[HKEY * PITCH];
  __shared__ __bf16 sVt[4][DHEAD * PITCH];   // V transposed, per 32-key panel
  __shared__ __bf16 sP[4][WWIN * PITCH];     // softmax probs as A-fragments
  __shared__ float  smask[HKEY];
  const int tid = threadIdx.x;
  const int w = blockIdx.x >> 2;
  const int h = blockIdx.x & 3;
#pragma unroll
  for (int it = 0; it < 4; ++it) {           // Q: 32 x 32
    const int slot = it * 64 + tid;
    const int rr = slot >> 3, c4 = (slot & 7) << 2;
    const float4 f = *(const float4*)(qh + (size_t)(w * WWIN + rr) * DMODEL + h * DHEAD + c4);
    __bf16* d = sQ + rr * PITCH + c4;
    d[0] = (__bf16)f.x; d[1] = (__bf16)f.y; d[2] = (__bf16)f.z; d[3] = (__bf16)f.w;
  }
#pragma unroll
  for (int it = 0; it < 16; ++it) {          // K: 128 x 32
    const int slot = it * 64 + tid;
    const int rr = slot >> 3, c4 = (slot & 7) << 2;
    const float4 f = *(const float4*)(kh + (size_t)(w * HKEY + rr) * DMODEL + h * DHEAD + c4);
    __bf16* d = sK + rr * PITCH + c4;
    d[0] = (__bf16)f.x; d[1] = (__bf16)f.y; d[2] = (__bf16)f.z; d[3] = (__bf16)f.w;
  }
#pragma unroll
  for (int it = 0; it < 16; ++it) {          // V: 128 x 32, transposed into [dh][key]
    const int slot = it * 64 + tid;
    const int key = slot >> 3, d0 = (slot & 7) << 2;
    const float4 f = *(const float4*)(vh + (size_t)(w * HKEY + key) * DMODEL + h * DHEAD + d0);
    const int kc = key >> 5, cc = key & 31;
    sVt[kc][(d0 + 0) * PITCH + cc] = (__bf16)f.x;
    sVt[kc][(d0 + 1) * PITCH + cc] = (__bf16)f.y;
    sVt[kc][(d0 + 2) * PITCH + cc] = (__bf16)f.z;
    sVt[kc][(d0 + 3) * PITCH + cc] = (__bf16)f.w;
  }
#pragma unroll
  for (int it = 0; it < 2; ++it) {           // key mask
    const int k = it * 64 + tid;
    smask[k] = pad[kidx[w * HKEY + k]] ? 0.0f : -1e9f;
  }
  __syncthreads();
  const int t    = tid >> 5;                 // wave -> query m-tile (16 rows)
  const int lane = tid & 31;
  const int lo   = lane & 15;
  const int hi8  = (lane >> 4) << 3;
  // logits = Q @ K^T : 16x128 per wave, 8 WMMAs
  v8f accL[8];
  v16bf aq = load_frag(sQ, t * 16, PITCH);
#pragma unroll
  for (int j = 0; j < 8; ++j) {
    v8f z = {};
    v16bf bk = load_frag(sK, j * 16, PITCH);
    accL[j] = __builtin_amdgcn_wmma_f32_16x16x32_bf16(false, aq, false, bk, (short)0, z, false, false);
  }
  const float scale = 0.17677669529663687f;  // 1/sqrt(32)
  const float* bp = bias + ((size_t)(w * NHEAD + h) * WWIN) * HKEY;
#pragma unroll
  for (int v = 0; v < 8; ++v) {              // softmax per query row (in registers)
    const int qrow = t * 16 + hi8 + v;
    float mx = -3.0e38f;
#pragma unroll
    for (int j = 0; j < 8; ++j) {
      const int key = j * 16 + lo;
      float x = accL[j][v] * scale + bp[(size_t)qrow * HKEY + key];
      x = (smask[key] == 0.0f) ? x : -1e9f;
      accL[j][v] = x;
      mx = fmaxf(mx, x);
    }
    for (int m = 8; m; m >>= 1) mx = fmaxf(mx, __shfl_xor(mx, m));
    float sum = 0.0f;
#pragma unroll
    for (int j = 0; j < 8; ++j) {
      const float e = __expf(accL[j][v] - mx);
      accL[j][v] = e; sum += e;
    }
    for (int m = 8; m; m >>= 1) sum += __shfl_xor(sum, m);
    const float rs = 1.0f / sum;
#pragma unroll
    for (int j = 0; j < 8; ++j) {
      const int key = j * 16 + lo;
      sP[key >> 5][qrow * PITCH + (key & 31)] = (__bf16)(accL[j][v] * rs);
    }
  }
  __syncthreads();
  // O = P @ V : 16x32 per wave, 2x4 WMMAs; epilogue fuses g * o
  v8f accO[2] = {};
#pragma unroll
  for (int kc = 0; kc < 4; ++kc) {
    v16bf ap = load_frag(sP[kc], t * 16, PITCH);
#pragma unroll
    for (int jn = 0; jn < 2; ++jn) {
      v16bf bv = load_frag(sVt[kc], jn * 16, PITCH);
      accO[jn] = __builtin_amdgcn_wmma_f32_16x16x32_bf16(false, ap, false, bv,
                                                         (short)0, accO[jn], false, false);
    }
  }
#pragma unroll
  for (int jn = 0; jn < 2; ++jn) {
    const int col = h * DHEAD + jn * 16 + lo;
#pragma unroll
    for (int v = 0; v < 8; ++v) {
      const int row = w * WWIN + t * 16 + hi8 + v;
      const size_t off = (size_t)row * DMODEL + col;
      o[off] = g[off] * accO[jn][v];
    }
  }
}

// ---------------- deterministic token mean pooling (one block per token) ----------------
__global__ __launch_bounds__(256) void k_pool(const float* __restrict__ qa,
                                              const int* __restrict__ ids,
                                              float* __restrict__ out) {
  __shared__ int sid[256];
  const int t = blockIdx.x;
  const int tid = threadIdx.x;
  float a0 = 0.f, a1 = 0.f, a2 = 0.f, cnt = 0.f;
  for (int base = 0; base < NATOM; base += 256) {
#if defined(HAVE_ASYNC_LDS)
    __builtin_amdgcn_global_load_async_to_lds_b32(
        (gint_t*)(ids + base + tid), (lint_t*)&sid[tid], 0, 0);
#if __has_builtin(__builtin_amdgcn_s_wait_asynccnt)
    __builtin_amdgcn_s_wait_asynccnt(0);
#else
    asm volatile("s_wait_asynccnt 0" ::: "memory");
#endif
#else
    sid[tid] = ids[base + tid];
#endif
    __syncthreads();
    for (int i = 0; i < 256; ++i) {
      if (sid[i] == t) {
        const float* row = qa + (size_t)(base + i) * FTOK;
        a0 += row[tid]; a1 += row[tid + 256]; a2 += row[tid + 512];
        cnt += 1.0f;
      }
    }
    __syncthreads();
  }
  const float s = 1.0f / (cnt + 1e-6f);
  out[(size_t)t * FTOK + tid]       = a0 * s;
  out[(size_t)t * FTOK + tid + 256] = a1 * s;
  out[(size_t)t * FTOK + tid + 512] = a2 * s;
}

// ---------------- host orchestration ----------------
extern "C" void kernel_launch(void* const* d_in, const int* in_sizes, int n_in,
                              void* d_out, int out_size, void* d_ws, size_t ws_size,
                              hipStream_t stream) {
  (void)in_sizes; (void)n_in; (void)out_size; (void)ws_size;
  const float* q         = (const float*)d_in[0];
  const float* c         = (const float*)d_in[1];
  const float* r         = (const float*)d_in[2];
  const float* attn_bias = (const float*)d_in[3];
  const int*   token_ids = (const int*)d_in[4];
  const int*   key_idx   = (const int*)d_in[5];
  const unsigned char* pad = (const unsigned char*)d_in[6];
  const float* r_to_q_w  = (const float*)d_in[7];
  const float* adaln1_s  = (const float*)d_in[8];
  const float* adaln1_sh = (const float*)d_in[9];
  const float* wq        = (const float*)d_in[10];
  const float* bq        = (const float*)d_in[11];
  const float* wk        = (const float*)d_in[12];
  const float* wv        = (const float*)d_in[13];
  const float* wg        = (const float*)d_in[14];
  const float* wo        = (const float*)d_in[15];
  const float* wop       = (const float*)d_in[16];
  const float* bop       = (const float*)d_in[17];
  const float* adaln2_s  = (const float*)d_in[18];
  const float* adaln2_sh = (const float*)d_in[19];
  const float* w_gate    = (const float*)d_in[20];
  const float* w_lin     = (const float*)d_in[21];
  const float* w2        = (const float*)d_in[22];
  const float* wg2       = (const float*)d_in[23];
  const float* w_a2t     = (const float*)d_in[24];

  char* wp = (char*)d_ws;
  auto alloc = [&](size_t bytes) -> void* {
    void* ret = (void*)wp;
    wp += (bytes + 255) & ~(size_t)255;
    return ret;
  };
  const size_t ND = (size_t)NATOM * DMODEL;
  float* a     = (float*)alloc(ND * 4);
  float* c_n   = (float*)alloc(ND * 4);
  float* ln_a  = (float*)alloc(ND * 4);
  float* bbuf  = (float*)alloc(ND * 4);
  float* qhbuf = (float*)alloc(ND * 4);
  float* gbuf  = (float*)alloc(ND * 4);
  float* t0    = (float*)alloc(ND * 4);
  float* obuf  = (float*)alloc(ND * 4);
  float* hbuf  = (float*)alloc((size_t)NATOM * 256 * 4);
  float* khbuf = (float*)alloc((size_t)NWIN * HKEY * DMODEL * 4);
  float* vhbuf = (float*)alloc((size_t)NWIN * HKEY * DMODEL * 4);
  float* qabuf = (float*)alloc((size_t)NATOM * FTOK * 4);

  const int DD = DMODEL * DMODEL;
  __bf16* b_a1s   = (__bf16*)alloc((size_t)NLAYER * DD * 2);
  __bf16* b_a1sh  = (__bf16*)alloc((size_t)NLAYER * DD * 2);
  __bf16* b_wq    = (__bf16*)alloc((size_t)NLAYER * DD * 2);
  __bf16* b_wk    = (__bf16*)alloc((size_t)NLAYER * DD * 2);
  __bf16* b_wv    = (__bf16*)alloc((size_t)NLAYER * DD * 2);
  __bf16* b_wg    = (__bf16*)alloc((size_t)NLAYER * DD * 2);
  __bf16* b_wo    = (__bf16*)alloc((size_t)NLAYER * DD * 2);
  __bf16* b_wop   = (__bf16*)alloc((size_t)NLAYER * DD * 2);
  __bf16* b_a2s   = (__bf16*)alloc((size_t)NLAYER * DD * 2);
  __bf16* b_a2sh  = (__bf16*)alloc((size_t)NLAYER * DD * 2);
  __bf16* b_wg2   = (__bf16*)alloc((size_t)NLAYER * DD * 2);
  __bf16* b_wgate = (__bf16*)alloc((size_t)NLAYER * DMODEL * 256 * 2);
  __bf16* b_wlin  = (__bf16*)alloc((size_t)NLAYER * DMODEL * 256 * 2);
  __bf16* b_w2    = (__bf16*)alloc((size_t)NLAYER * 256 * DMODEL * 2);
  __bf16* b_a2t   = (__bf16*)alloc((size_t)DMODEL * FTOK * 2);

  auto cvt = [&](const float* s, __bf16* d, int n) {
    k_cvt<<<(n + 255) / 256, 256, 0, stream>>>(s, d, n);
  };
  cvt(adaln1_s,  b_a1s,   NLAYER * DD);
  cvt(adaln1_sh, b_a1sh,  NLAYER * DD);
  cvt(wq,        b_wq,    NLAYER * DD);
  cvt(wk,        b_wk,    NLAYER * DD);
  cvt(wv,        b_wv,    NLAYER * DD);
  cvt(wg,        b_wg,    NLAYER * DD);
  cvt(wo,        b_wo,    NLAYER * DD);
  cvt(wop,       b_wop,   NLAYER * DD);
  cvt(adaln2_s,  b_a2s,   NLAYER * DD);
  cvt(adaln2_sh, b_a2sh,  NLAYER * DD);
  cvt(wg2,       b_wg2,   NLAYER * DD);
  cvt(w_gate,    b_wgate, NLAYER * DMODEL * 256);
  cvt(w_lin,     b_wlin,  NLAYER * DMODEL * 256);
  cvt(w2,        b_w2,    NLAYER * 256 * DMODEL);
  cvt(w_a2t,     b_a2t,   DMODEL * FTOK);

  k_init_a<<<(NATOM * DMODEL) / 256, 256, 0, stream>>>(q, r, r_to_q_w, a);
  k_ln<<<NATOM / 4, 128, 0, stream>>>(c, c_n);

  const dim3 g128(NATOM / BM, DMODEL / BN);          // 64 x 2
  const dim3 g256(NATOM / BM, 256 / BN);             // 64 x 4
  const dim3 gkv((NWIN * HKEY) / BM, DMODEL / BN);   // 256 x 2
  const dim3 g768(NATOM / BM, FTOK / BN);            // 64 x 12

  for (int l = 0; l < NLAYER; ++l) {
    k_ln<<<NATOM / 4, 128, 0, stream>>>(a, ln_a);
    k_gemm_dual<DMODEL><<<g128, 256, 0, stream>>>(c_n, b_a1s + (size_t)l * DD,
                                                  b_a1sh + (size_t)l * DD, ln_a, bbuf,
                                                  DMODEL, DUAL_ADALN);
    k_gemm<DMODEL, false><<<g128, 256, 0, stream>>>(bbuf, nullptr, b_wq + (size_t)l * DD,
                                                    bq + l * DMODEL, qhbuf, nullptr, nullptr,
                                                    DMODEL, ACT_NONE);
    k_gemm<DMODEL, false><<<g128, 256, 0, stream>>>(bbuf, nullptr, b_wg + (size_t)l * DD,
                                                    nullptr, gbuf, nullptr, nullptr,
                                                    DMODEL, ACT_SIGMOID);
    k_gemm<DMODEL, true><<<gkv, 256, 0, stream>>>(bbuf, key_idx, b_wk + (size_t)l * DD,
                                                  nullptr, khbuf, nullptr, nullptr,
                                                  DMODEL, ACT_NONE);
    k_gemm<DMODEL, true><<<gkv, 256, 0, stream>>>(bbuf, key_idx, b_wv + (size_t)l * DD,
                                                  nullptr, vhbuf, nullptr, nullptr,
                                                  DMODEL, ACT_NONE);
    k_attn<<<NWIN * NHEAD, 64, 0, stream>>>(qhbuf, khbuf, vhbuf, gbuf, attn_bias,
                                            key_idx, pad, obuf);
    k_gemm<DMODEL, false><<<g128, 256, 0, stream>>>(c, nullptr, b_wop + (size_t)l * DD,
                                                    bop + l * DMODEL, t0, nullptr, nullptr,
                                                    DMODEL, ACT_SIGMOID);
    k_gemm<DMODEL, false><<<g128, 256, 0, stream>>>(obuf, nullptr, b_wo + (size_t)l * DD,
                                                    nullptr, nullptr, t0, a,
                                                    DMODEL, ACT_GATED);
    k_ln<<<NATOM / 4, 128, 0, stream>>>(a, ln_a);
    k_gemm_dual<DMODEL><<<g128, 256, 0, stream>>>(c_n, b_a2s + (size_t)l * DD,
                                                  b_a2sh + (size_t)l * DD, ln_a, bbuf,
                                                  DMODEL, DUAL_ADALN);
    k_gemm_dual<DMODEL><<<g256, 256, 0, stream>>>(bbuf, b_wgate + (size_t)l * DMODEL * 256,
                                                  b_wlin + (size_t)l * DMODEL * 256, nullptr,
                                                  hbuf, 256, DUAL_SWIGLU);
    k_gemm<DMODEL, false><<<g128, 256, 0, stream>>>(c, nullptr, b_wg2 + (size_t)l * DD,
                                                    nullptr, t0, nullptr, nullptr,
                                                    DMODEL, ACT_SIGMOID);
    k_gemm<256, false><<<g128, 256, 0, stream>>>(hbuf, nullptr, b_w2 + (size_t)l * 256 * DMODEL,
                                                 nullptr, nullptr, t0, a,
                                                 DMODEL, ACT_GATED);
  }
  k_gemm<DMODEL, false><<<g768, 256, 0, stream>>>(a, nullptr, b_a2t, nullptr, qabuf,
                                                  nullptr, nullptr, FTOK, ACT_RELU);
  k_pool<<<NTOK, 256, 0, stream>>>(qabuf, token_ids, (float*)d_out);
}
